// FastCoreAttention_27247272526206
// MI455X (gfx1250) — matrix operand order repaired and checked by reading, not compiled, not run on hardware
//
#include <hip/hip_runtime.h>
#include <cstdint>
#include <cstddef>

// ---------------------------------------------------------------------------
// HyperAttention (LSH sorted block attention + sampled residual) for MI455X.
// bf16 WMMA (v_wmma_f32_16x16x32_bf16) for QK^T and PV, fp32 softmax/merge.
// Register double-buffered B-operand loads to overlap VMEM with WMMA.
// ---------------------------------------------------------------------------

typedef __attribute__((ext_vector_type(16))) __bf16 v16bf;
typedef __attribute__((ext_vector_type(8)))  float  v8f;

union BF16x16 { v16bf v; uint4 u[2]; };

#define NHEADS   32
#define SEQLEN   8192
#define HDIM     128
#define BLOCK    256
#define NSAMP    256
#define NPROJ    7

__device__ __forceinline__ v8f wmma_bf16(const v16bf& a, const v16bf& b, const v8f& c) {
  return __builtin_amdgcn_wmma_f32_16x16x32_bf16(false, a, false, b, (short)0, c, false, false);
}

// ---------------------------------------------------------------------------
// 1) LSH hash: bucket[h][n] = gray(bits(x . proj))   (gray(i) = i ^ (i>>1))
// ---------------------------------------------------------------------------
__global__ void lsh_hash_kernel(const float* __restrict__ x,      // [SEQ][1][H][D]
                                const float* __restrict__ proj,   // [D][NPROJ]
                                int* __restrict__ bucket) {       // [H][SEQ]
  __shared__ float sproj[HDIM * NPROJ];
  for (int i = threadIdx.x; i < HDIM * NPROJ; i += blockDim.x) sproj[i] = proj[i];
  __syncthreads();

  int idx = blockIdx.x * blockDim.x + threadIdx.x;   // idx = n*32 + h
  int n = idx >> 5, h = idx & 31;
  const float* xp = x + (size_t)idx * HDIM;          // ((n*1+0)*32+h)*128
  float s[NPROJ];
#pragma unroll
  for (int r = 0; r < NPROJ; r++) s[r] = 0.0f;
  for (int d = 0; d < HDIM; d++) {
    float xv = xp[d];
#pragma unroll
    for (int r = 0; r < NPROJ; r++) s[r] += xv * sproj[d * NPROJ + r];
  }
  int bits = 0;
#pragma unroll
  for (int r = 0; r < NPROJ; r++) bits |= (s[r] > 0.0f) ? (1 << r) : 0;
  bucket[h * SEQLEN + n] = bits ^ (bits >> 1);
}

// ---------------------------------------------------------------------------
// 2) Stable counting sort per head: 128 buckets, 256 threads x 32-token chunks
//    dynamic LDS: hist[128][256] u32 + basePS[128]
// ---------------------------------------------------------------------------
__global__ void bucket_sort_kernel(const int* __restrict__ bucket,  // [H][SEQ]
                                   int* __restrict__ sortidx) {     // [H][SEQ] sorted pos -> orig
  extern __shared__ unsigned int sm[];
  unsigned int* hist  = sm;             // 128*256
  unsigned int* baseP = sm + 128 * 256; // 128
  const int h = blockIdx.x;
  const int t = threadIdx.x;            // 0..255
  const int* bk = bucket + h * SEQLEN + t * 32;

  for (int v = 0; v < 128; v++) hist[v * 256 + t] = 0;
  __syncthreads();
  for (int j = 0; j < 32; j++) hist[bk[j] * 256 + t]++;
  __syncthreads();
  if (t < 128) {                        // exclusive scan over threads, per bucket
    unsigned int run = 0;
    for (int i = 0; i < 256; i++) { unsigned int c = hist[t * 256 + i]; hist[t * 256 + i] = run; run += c; }
    baseP[t] = run;
  }
  __syncthreads();
  if (t == 0) {                         // exclusive scan over buckets
    unsigned int run = 0;
    for (int v = 0; v < 128; v++) { unsigned int c = baseP[v]; baseP[v] = run; run += c; }
  }
  __syncthreads();
  for (int j = 0; j < 32; j++) {        // stable scatter
    int v = bk[j];
    unsigned int r = baseP[v] + hist[v * 256 + t];
    hist[v * 256 + t]++;
    sortidx[h * SEQLEN + r] = t * 32 + j;
  }
}

// ---------------------------------------------------------------------------
// 3) Gather sorted Q/K (row-major bf16) and V transposed [h][d][key] bf16
// ---------------------------------------------------------------------------
__global__ void gather_qkv_kernel(const float* __restrict__ q, const float* __restrict__ k,
                                  const float* __restrict__ v,
                                  const int* __restrict__ sortq, const int* __restrict__ sortk,
                                  __bf16* __restrict__ Qs, __bf16* __restrict__ Ks,
                                  __bf16* __restrict__ Vt) {
  int idx = blockIdx.x * blockDim.x + threadIdx.x;   // h * SEQ * D elements
  int d = idx & 127;
  int p = (idx >> 7) & (SEQLEN - 1);
  int h = idx >> 20;
  int oq = sortq[h * SEQLEN + p];
  int ok = sortk[h * SEQLEN + p];
  Qs[(size_t)(h * SEQLEN + p) * HDIM + d] = (__bf16)q[((size_t)oq * 32 + h) * HDIM + d];
  Ks[(size_t)(h * SEQLEN + p) * HDIM + d] = (__bf16)k[((size_t)ok * 32 + h) * HDIM + d];
  Vt[((size_t)h * HDIM + d) * SEQLEN + p] = (__bf16)v[((size_t)ok * 32 + h) * HDIM + d];
}

// ---------------------------------------------------------------------------
// 4) Gather sampled residual keys/values (indices index into the SORTED arrays)
// ---------------------------------------------------------------------------
__global__ void gather_sampled_kernel(const __bf16* __restrict__ Ks, const __bf16* __restrict__ Vt,
                                      const int* __restrict__ samp,   // [1][H][NSAMP]
                                      __bf16* __restrict__ Ksub,      // [H][NSAMP][D]
                                      __bf16* __restrict__ Vsubt,     // [H][D][NSAMP]
                                      int* __restrict__ colblk) {     // [H][NSAMP]
  int idx = blockIdx.x * blockDim.x + threadIdx.x;   // h * NSAMP * D elements
  int d = idx & 127;
  int j = (idx >> 7) & (NSAMP - 1);
  int h = idx >> 15;
  int s = samp[h * NSAMP + j];
  Ksub[((size_t)h * NSAMP + j) * HDIM + d] = Ks[((size_t)h * SEQLEN + s) * HDIM + d];
  Vsubt[((size_t)h * HDIM + d) * NSAMP + j] = Vt[((size_t)h * HDIM + d) * SEQLEN + s];
  if (d == 0) colblk[h * NSAMP + j] = s >> 8;
}

// ---------------------------------------------------------------------------
// 5) Attention: one workgroup (4 waves) per (head, 16-row q-tile).
//    Waves 0,1: block-diagonal keys (128 each). Waves 2,3: sampled keys.
//    Per-wave flash partial (m, l, O); residual m += ln(n/S); LDS merge.
//    B operands are double-buffered in registers so next tile's global_load_b128s
//    are in flight while the current tile's WMMAs execute.
// ---------------------------------------------------------------------------
__global__ void attention_kernel(const __bf16* __restrict__ Qs, const __bf16* __restrict__ Ks,
                                 const __bf16* __restrict__ Vt,
                                 const __bf16* __restrict__ Ksub, const __bf16* __restrict__ Vsubt,
                                 const int* __restrict__ colblk, const int* __restrict__ sortq,
                                 float* __restrict__ out) {
  constexpr float SCALE = 0.08838834764831845f;   // 1/sqrt(128)
  constexpr float LN_F  = 3.4657359027997265f;    // ln(8192/256)
  extern __shared__ char smraw[];
  float*  sS   = (float*)smraw;                   // [4][16][128] fp32  (32KB)
  __bf16* sP   = (__bf16*)(smraw + 32768);        // [4][16][128] bf16  (16KB)
  float*  sO   = (float*)(smraw + 49152);         // [4][16][128] fp32  (32KB)
  float*  sM   = (float*)(smraw + 81920);         // [4][16]
  float*  sL   = sM + 64;                         // [4][16]
  float*  sRed = sL + 64;                         // [4][32]

  const int qtile = blockIdx.x;                   // 0..511
  const int h     = blockIdx.y;                   // 0..31
  const int row0  = qtile * 16;
  const int rowblk = qtile >> 4;                  // sorted block index of this tile
  const int wave = threadIdx.x >> 5;
  const int lane = threadIdx.x & 31;
  const int ln   = lane & 15;                     // N-col / M-row lane id
  const int lh   = lane >> 4;                     // lane half

  // ---- Q tile as 4 A-operands (K = d, 32 per operand) -----------------------
  const __bf16* qrow = Qs + ((size_t)h * SEQLEN + row0 + ln) * HDIM;
  BF16x16 qa[4];
#pragma unroll
  for (int kc = 0; kc < 4; kc++) {
    qa[kc].u[0] = *(const uint4*)(qrow + kc * 32 + lh * 16);
    qa[kc].u[1] = *(const uint4*)(qrow + kc * 32 + lh * 16 + 8);
  }

  // ---- key sources per wave -------------------------------------------------
  const bool resid = (wave >= 2);
  const int  key0  = resid ? (wave - 2) * 128 : rowblk * 256 + wave * 128;
  const __bf16* kbase = resid ? (Ksub + (size_t)h * NSAMP * HDIM)
                              : (Ks   + (size_t)h * SEQLEN * HDIM);

  // ---- S = scale * Q K^T (+ mask) -> LDS, double-buffered K operands --------
  {
    BF16x16 kb[4];
    const __bf16* krow0 = kbase + (size_t)(key0 + ln) * HDIM;
#pragma unroll
    for (int kc = 0; kc < 4; kc++) {
      kb[kc].u[0] = *(const uint4*)(krow0 + kc * 32 + lh * 16);
      kb[kc].u[1] = *(const uint4*)(krow0 + kc * 32 + lh * 16 + 8);
    }
    for (int kt = 0; kt < 8; kt++) {
      BF16x16 kbn[4];
      if (kt < 7) {                                // issue next tile's loads first
        const __bf16* krow = kbase + (size_t)(key0 + (kt + 1) * 16 + ln) * HDIM;
        __builtin_prefetch(krow + 2 * HDIM, 0, 3);
#pragma unroll
        for (int kc = 0; kc < 4; kc++) {
          kbn[kc].u[0] = *(const uint4*)(krow + kc * 32 + lh * 16);
          kbn[kc].u[1] = *(const uint4*)(krow + kc * 32 + lh * 16 + 8);
        }
      }
      v8f acc = {0.f, 0.f, 0.f, 0.f, 0.f, 0.f, 0.f, 0.f};
#pragma unroll
      for (int kc = 0; kc < 4; kc++) acc = wmma_bf16(qa[kc].v, kb[kc].v, acc);
      float mv = 0.0f;
      if (resid) {
        int j = key0 + kt * 16 + ln;
        mv = (colblk[h * NSAMP + j] == rowblk) ? -3.0e38f : 0.0f;
      }
#pragma unroll
      for (int j = 0; j < 8; j++)
        sS[wave * 2048 + (j + lh * 8) * 128 + kt * 16 + ln] = acc[j] * SCALE + mv;
#pragma unroll
      for (int kc = 0; kc < 4; kc++) kb[kc] = kbn[kc];
    }
  }
  __syncthreads();

  // ---- row softmax stats + bf16 P ------------------------------------------
  {
    const float* srow = sS + wave * 2048 + ln * 128 + lh * 64;
    float mpart = -3.4e38f;
    for (int c = 0; c < 64; c++) mpart = fmaxf(mpart, srow[c]);
    sRed[wave * 32 + lh * 16 + ln] = mpart;
    __syncthreads();
    float m = fmaxf(sRed[wave * 32 + ln], sRed[wave * 32 + 16 + ln]);
    __bf16* prow = sP + wave * 2048 + ln * 128 + lh * 64;
    float lpart = 0.0f;
    for (int c = 0; c < 64; c++) {
      float p = __expf(srow[c] - m);
      lpart += p;
      prow[c] = (__bf16)p;
    }
    __syncthreads();
    sRed[wave * 32 + lh * 16 + ln] = lpart;
    __syncthreads();
    if (lh == 0) {
      sM[wave * 16 + ln] = m + (resid ? LN_F : 0.0f);
      sL[wave * 16 + ln] = sRed[wave * 32 + ln] + sRed[wave * 32 + 16 + ln];
    }
  }

  // ---- O_partial = P V  (V transposed: [d][key]), double-buffered V ---------
  const __bf16* vbase = resid ? (Vsubt + (size_t)h * HDIM * NSAMP)
                              : (Vt    + (size_t)h * HDIM * SEQLEN);
  const int vstride = resid ? NSAMP : SEQLEN;
  v8f o[8];
#pragma unroll
  for (int dt = 0; dt < 8; dt++) o[dt] = (v8f){0.f, 0.f, 0.f, 0.f, 0.f, 0.f, 0.f, 0.f};
#pragma unroll
  for (int kc = 0; kc < 4; kc++) {
    const __bf16* pr = sP + wave * 2048 + ln * 128 + kc * 32 + lh * 16;
    BF16x16 pa;
    pa.u[0] = *(const uint4*)pr;
    pa.u[1] = *(const uint4*)(pr + 8);
    BF16x16 vb;
    {
      const __bf16* vr = vbase + (size_t)ln * vstride + key0 + kc * 32 + lh * 16;
      vb.u[0] = *(const uint4*)vr;
      vb.u[1] = *(const uint4*)(vr + 8);
    }
#pragma unroll
    for (int dt = 0; dt < 8; dt++) {
      BF16x16 vbn;
      if (dt < 7) {                                // next d-tile's loads in flight
        const __bf16* vr = vbase + (size_t)((dt + 1) * 16 + ln) * vstride + key0 + kc * 32 + lh * 16;
        vbn.u[0] = *(const uint4*)vr;
        vbn.u[1] = *(const uint4*)(vr + 8);
      }
      o[dt] = wmma_bf16(pa.v, vb.v, o[dt]);
      vb = vbn;
    }
  }
#pragma unroll
  for (int dt = 0; dt < 8; dt++)
#pragma unroll
    for (int j = 0; j < 8; j++)
      sO[wave * 2048 + (j + lh * 8) * 128 + dt * 16 + ln] = o[dt][j];
  __syncthreads();

  // ---- merge 4 partials (flash merge; residual weight folded into m) --------
  {
    const int tid = threadIdx.x;
    const int row = tid >> 3;          // 16 rows
    const int cg  = tid & 7;           // 8 col-groups of 16
    float mm[4], ll[4], M = -3.4e38f;
#pragma unroll
    for (int w = 0; w < 4; w++) { mm[w] = sM[w * 16 + row]; ll[w] = sL[w * 16 + row]; M = fmaxf(M, mm[w]); }
    float wgt[4], den = 0.0f;
#pragma unroll
    for (int w = 0; w < 4; w++) { wgt[w] = __expf(mm[w] - M); den += wgt[w] * ll[w]; }
    const float inv = 1.0f / den;
    const int orig = sortq[h * SEQLEN + row0 + row];
    float* op = out + (size_t)orig * (NHEADS * HDIM) + h * HDIM + cg * 16;
    for (int c = 0; c < 16; c++) {
      float num = 0.0f;
#pragma unroll
      for (int w = 0; w < 4; w++) num += wgt[w] * sO[w * 2048 + row * 128 + cg * 16 + c];
      op[c] = num * inv;
    }
  }
}

// ---------------------------------------------------------------------------
// launch
// ---------------------------------------------------------------------------
extern "C" void kernel_launch(void* const* d_in, const int* in_sizes, int n_in,
                              void* d_out, int out_size, void* d_ws, size_t ws_size,
                              hipStream_t stream) {
  (void)in_sizes; (void)n_in; (void)out_size; (void)ws_size;
  const float* q    = (const float*)d_in[0];
  const float* k    = (const float*)d_in[1];
  const float* v    = (const float*)d_in[2];
  const float* proj = (const float*)d_in[3];
  const int*   samp = (const int*)d_in[4];
  float* out = (float*)d_out;

  char* ws = (char*)d_ws;
  const size_t MB = 1ull << 20;
  int*    bq    = (int*)(ws + 0 * MB);     // [H][SEQ]
  int*    bk    = (int*)(ws + 1 * MB);
  int*    sq    = (int*)(ws + 2 * MB);
  int*    sk    = (int*)(ws + 3 * MB);
  __bf16* Qs    = (__bf16*)(ws + 4 * MB);   // 64 MB
  __bf16* Ks    = (__bf16*)(ws + 68 * MB);  // 64 MB
  __bf16* Vt    = (__bf16*)(ws + 132 * MB); // 64 MB
  __bf16* Ksub  = (__bf16*)(ws + 196 * MB); // 2 MB
  __bf16* Vsubt = (__bf16*)(ws + 198 * MB); // 2 MB
  int*    cblk  = (int*)(ws + 200 * MB);    // 32 KB

  // 1) LSH hash q and k
  lsh_hash_kernel<<<SEQLEN * NHEADS / 256, 256, 0, stream>>>(q, proj, bq);
  lsh_hash_kernel<<<SEQLEN * NHEADS / 256, 256, 0, stream>>>(k, proj, bk);

  // 2) stable counting sort per head
  const size_t sort_lds = (128 * 256 + 128) * sizeof(unsigned int);
  bucket_sort_kernel<<<NHEADS, 256, sort_lds, stream>>>(bq, sq);
  bucket_sort_kernel<<<NHEADS, 256, sort_lds, stream>>>(bk, sk);

  // 3) gather sorted Q/K row-major, V transposed (bf16)
  gather_qkv_kernel<<<(NHEADS * SEQLEN * HDIM) / 256, 256, 0, stream>>>(q, k, v, sq, sk, Qs, Ks, Vt);

  // 4) gather sampled residual keys
  gather_sampled_kernel<<<(NHEADS * NSAMP * HDIM) / 256, 256, 0, stream>>>(Ks, Vt, samp, Ksub, Vsubt, cblk);

  // 5) block-diag + residual attention, merge, scatter to unsorted rows
  const size_t attn_lds = 81920 + (64 + 64 + 128) * sizeof(float);  // 82944 B
  attention_kernel<<<dim3(SEQLEN / 16, NHEADS), 128, attn_lds, stream>>>(
      Qs, Ks, Vt, Ksub, Vsubt, cblk, sq, out);
}
